// WLSMLPLayerE_49065706389960
// MI455X (gfx1250) — matrix operand order repaired and checked by reading, not compiled, not run on hardware
//
#include <hip/hip_runtime.h>
#include <hip/hip_bf16.h>
#include <math.h>

typedef __attribute__((ext_vector_type(16))) __bf16 v16bf;
typedef __attribute__((ext_vector_type(8)))  float  v8f;
typedef __attribute__((ext_vector_type(4)))  int    v4i;

#define BM 128
#define BN 64
#define BK 64
#define LDS_STRIDE (BK + 8)   // 72 elems = 144B: 16B-aligned, conflict-free frag reads

#define AS_GLOBAL __attribute__((address_space(1)))
#define AS_LDS    __attribute__((address_space(3)))

// 16-byte global -> LDS copy. Prefer the gfx1250 async LDS-direct path
// (GLOBAL_LOAD_ASYNC_TO_LDS_B128, ASYNCcnt) to avoid the VGPR round-trip.
__device__ __forceinline__ void async_copy16(const __bf16* g, __bf16* l) {
#if __has_builtin(__builtin_amdgcn_global_load_async_to_lds_b128)
    __builtin_amdgcn_global_load_async_to_lds_b128(
        (AS_GLOBAL v4i*)g, (AS_LDS v4i*)l, 0, 0);
#else
    *(uint4*)l = *(const uint4*)g;
#endif
}

__device__ __forceinline__ void wait_async0() {
#if __has_builtin(__builtin_amdgcn_s_wait_asynccnt)
    __builtin_amdgcn_s_wait_asynccnt(0);
#else
    asm volatile("s_wait_asynccnt 0x0" ::: "memory");
#endif
}

// A-fragment (16x32 bf16, M x K): lane<16 -> K {0..7,16..23}; lane>=16 -> K {8..15,24..31}
__device__ __forceinline__ v16bf load_fragA(const __bf16* base, int row, int ks, int lhi) {
    union { v16bf v; uint4 q[2]; } f;
    const __bf16* p = base + row * LDS_STRIDE + ks + 8 * lhi;
    f.q[0] = *(const uint4*)(p);
    f.q[1] = *(const uint4*)(p + 16);
    return f.v;
}

// B-fragment (32x16 bf16, K x N, stored N-major in LDS): lane<16 -> K {0..7,16..23}; lane>=16 -> K {8..15,24..31}
__device__ __forceinline__ v16bf load_fragB(const __bf16* base, int row, int ks, int lhi) {
    union { v16bf v; uint4 q[2]; } f;
    const __bf16* p = base + row * LDS_STRIDE + ks + 8 * lhi;
    f.q[0] = *(const uint4*)(p);
    f.q[1] = *(const uint4*)(p + 16);
    return f.v;
}

// C[M,N] = act(A_bf16[M,K] @ Wt_bf16[N,K]^T + b) ; K,N multiples of 64.
// Wt is pre-transposed (N-major) bf16 so both stagings are pure 16B async copies.
// Optional f32 output (stride ldF) and/or bf16 output (stride N).
__global__ __launch_bounds__(256)
void k_gemm_bf16(const __bf16* __restrict__ A, const __bf16* __restrict__ Wt,
                 const float* __restrict__ bias,
                 float* outF, int ldF, __bf16* outB,
                 int M, int N, int K, int relu)
{
    __shared__ __bf16 As[BM * LDS_STRIDE];
    __shared__ __bf16 Bs[BN * LDS_STRIDE];

    const int tid  = threadIdx.x;
    const int m0   = blockIdx.x * BM;
    const int n0   = blockIdx.y * BN;
    const int wave = tid >> 5;
    const int lane = tid & 31;
    const int lrow = lane & 15;
    const int lhi  = lane >> 4;
    const int wm   = (wave & 3) * 32;   // wave M offset within block tile
    const int wn   = (wave >> 2) * 32;  // wave N offset within block tile

    v8f acc[2][2];
    #pragma unroll
    for (int i = 0; i < 2; ++i)
        #pragma unroll
        for (int j = 0; j < 2; ++j)
            acc[i][j] = (v8f){0.f,0.f,0.f,0.f,0.f,0.f,0.f,0.f};

    for (int k0 = 0; k0 < K; k0 += BK) {
        // ---- stage A tile [BM x BK]: 1024 async 16B copies (rows clamped, no branches;
        //      rows >= M are garbage but only ever feed store-guarded output rows) ----
        #pragma unroll
        for (int it = 0; it < 4; ++it) {
            int c   = tid + it * 256;
            int row = c >> 3;
            int col = (c & 7) * 8;
            int gr  = m0 + row;
            if (gr > M - 1) gr = M - 1;
            async_copy16(A + (size_t)gr * K + k0 + col, &As[row * LDS_STRIDE + col]);
        }
        // ---- stage B tile [BN x BK] from pre-transposed bf16 weights: 512 async copies ----
        #pragma unroll
        for (int it = 0; it < 2; ++it) {
            int c   = tid + it * 256;
            int n   = c >> 3;
            int col = (c & 7) * 8;
            async_copy16(Wt + (size_t)(n0 + n) * K + k0 + col, &Bs[n * LDS_STRIDE + col]);
        }
        wait_async0();
        __syncthreads();

        #pragma unroll
        for (int ks = 0; ks < BK; ks += 32) {
            v16bf a0 = load_fragA(As, wm + 0,  ks, lhi);
            v16bf a1 = load_fragA(As, wm + 16, ks, lhi);
            v16bf b0 = load_fragB(Bs, wn + 0,  ks, lhi);
            v16bf b1 = load_fragB(Bs, wn + 16, ks, lhi);
            acc[0][0] = __builtin_amdgcn_wmma_f32_16x16x32_bf16(false, a0, false, b0, (short)0, acc[0][0], false, false);
            acc[0][1] = __builtin_amdgcn_wmma_f32_16x16x32_bf16(false, a0, false, b1, (short)0, acc[0][1], false, false);
            acc[1][0] = __builtin_amdgcn_wmma_f32_16x16x32_bf16(false, a1, false, b0, (short)0, acc[1][0], false, false);
            acc[1][1] = __builtin_amdgcn_wmma_f32_16x16x32_bf16(false, a1, false, b1, (short)0, acc[1][1], false, false);
        }
        __syncthreads();
    }

    // ---- epilogue: C/D layout: VGPR j holds row (j + 8*lhi), col = lane%16 ----
    #pragma unroll
    for (int mt = 0; mt < 2; ++mt) {
        #pragma unroll
        for (int nt = 0; nt < 2; ++nt) {
            int col   = n0 + wn + nt * 16 + lrow;
            float bv  = bias[col];
            int rbase = m0 + wm + mt * 16 + 8 * lhi;
            #pragma unroll
            for (int j = 0; j < 8; ++j) {
                int r = rbase + j;
                if (r < M) {
                    float v = acc[mt][nt][j] + bv;
                    if (relu) v = fmaxf(v, 0.f);
                    if (outF) outF[(size_t)r * ldF + col] = v;
                    if (outB) outB[(size_t)r * N   + col] = (__bf16)v;
                }
            }
        }
    }
}

__global__ void k_f32_to_bf16(const float* __restrict__ x, __bf16* __restrict__ y, int n4) {
    int i = blockIdx.x * blockDim.x + threadIdx.x;
    if (i >= n4) return;
    float4 f = ((const float4*)x)[i];
    union { __bf16 t[4]; uint2 q; } u;
    u.t[0] = (__bf16)f.x; u.t[1] = (__bf16)f.y; u.t[2] = (__bf16)f.z; u.t[3] = (__bf16)f.w;
    ((uint2*)y)[i] = u.q;
}

// one-time weight transpose+convert: W[k][n] f32 -> Wt[n][k] bf16
__global__ void k_wt_bf16(const float* __restrict__ W, __bf16* __restrict__ Wt, int K, int N) {
    int idx = blockIdx.x * blockDim.x + threadIdx.x;
    if (idx >= K * N) return;
    int k = idx / N;
    int n = idx - k * N;
    Wt[(size_t)n * K + k] = (__bf16)W[idx];
}

// zero out[:, 64:128] (agg half), 2 floats per thread
__global__ void k_zero_agg(float* __restrict__ out, int n2) {
    int i = blockIdx.x * blockDim.x + threadIdx.x;
    if (i >= n2) return;
    int row = i >> 5;
    int c   = (i & 31) * 2;
    *(float2*)(out + (size_t)row * 128 + 64 + c) = make_float2(0.f, 0.f);
}

// one wave (32 lanes) per edge: gather hs[src], hd[dst], dot -> sigmoid gate,
// scatter gated src-feature into out[dst, 64:128] with f32 atomics (L2-resident)
__global__ __launch_bounds__(256)
void k_edges(const float* __restrict__ hs, const float* __restrict__ hd,
             const int* __restrict__ src, const int* __restrict__ dst,
             float* __restrict__ out, int E)
{
    int wid  = blockIdx.x * 8 + (threadIdx.x >> 5);
    int lane = threadIdx.x & 31;
    if (wid >= E) return;
    int s = src[wid];
    int d = dst[wid];
    float2 a = *(const float2*)(hs + (size_t)s * 64 + lane * 2);
    float2 b = *(const float2*)(hd + (size_t)d * 64 + lane * 2);
    float p = a.x * b.x + a.y * b.y;
    #pragma unroll
    for (int off = 16; off > 0; off >>= 1)
        p += __shfl_xor(p, off, 32);
    float gate = 1.f / (1.f + __expf(-p * 0.125f));   // 1/sqrt(64) = 0.125
    float* o = out + (size_t)d * 128 + 64 + lane * 2;
    atomicAdd(o,     a.x * gate);
    atomicAdd(o + 1, a.y * gate);
}

extern "C" void kernel_launch(void* const* d_in, const int* in_sizes, int n_in,
                              void* d_out, int out_size, void* d_ws, size_t ws_size,
                              hipStream_t stream)
{
    const float* features = (const float*)d_in[0];
    const int*   src      = (const int*)d_in[1];
    const int*   dst      = (const int*)d_in[2];
    const float* W1  = (const float*)d_in[3];
    const float* b1  = (const float*)d_in[4];
    const float* W2  = (const float*)d_in[5];
    const float* b2  = (const float*)d_in[6];
    const float* W3  = (const float*)d_in[7];
    const float* b3  = (const float*)d_in[8];
    const float* Wsp = (const float*)d_in[9];
    const float* bsp = (const float*)d_in[10];
    const float* Wdp = (const float*)d_in[11];
    const float* bdp = (const float*)d_in[12];
    float* out = (float*)d_out;

    const int H1v = in_sizes[4];          // 256
    const int H2v = in_sizes[6];          // 512
    const int P   = in_sizes[8];          // 64
    const int IN  = in_sizes[3] / H1v;    // 128
    const int Nn  = in_sizes[0] / IN;     // 100000
    const int E   = in_sizes[1];          // 1600000

    // workspace partition (all 256B-aligned)
    char* ws = (char*)d_ws;
    size_t off = 0;
    auto alignup = [](size_t x) { return (x + 255) & ~(size_t)255; };
    __bf16* Xbf  = (__bf16*)(ws + off); off = alignup(off + (size_t)Nn * IN  * 2);
    __bf16* h1   = (__bf16*)(ws + off); off = alignup(off + (size_t)Nn * H1v * 2);
    __bf16* h2   = (__bf16*)(ws + off); off = alignup(off + (size_t)Nn * H2v * 2);
    __bf16* hbf  = (__bf16*)(ws + off); off = alignup(off + (size_t)Nn * P   * 2);
    float*  hs   = (float*)(ws + off);  off = alignup(off + (size_t)Nn * P   * 4);
    float*  hd   = (float*)(ws + off);  off = alignup(off + (size_t)Nn * P   * 4);
    __bf16* Wt1  = (__bf16*)(ws + off); off = alignup(off + (size_t)IN  * H1v * 2);
    __bf16* Wt2  = (__bf16*)(ws + off); off = alignup(off + (size_t)H1v * H2v * 2);
    __bf16* Wt3  = (__bf16*)(ws + off); off = alignup(off + (size_t)H2v * P   * 2);
    __bf16* Wtsp = (__bf16*)(ws + off); off = alignup(off + (size_t)P   * P   * 2);
    __bf16* Wtdp = (__bf16*)(ws + off); off = alignup(off + (size_t)P   * P   * 2);
    (void)ws_size; (void)n_in; (void)out_size;

    const int mblocks = (Nn + BM - 1) / BM;

    // one-time conversions: features f32->bf16, weights f32->bf16 transposed (N-major)
    {
        int n4 = (Nn * IN) / 4;
        k_f32_to_bf16<<<(n4 + 255) / 256, 256, 0, stream>>>(features, Xbf, n4);
    }
    k_wt_bf16<<<(IN  * H1v + 255) / 256, 256, 0, stream>>>(W1,  Wt1,  IN,  H1v);
    k_wt_bf16<<<(H1v * H2v + 255) / 256, 256, 0, stream>>>(W2,  Wt2,  H1v, H2v);
    k_wt_bf16<<<(H2v * P   + 255) / 256, 256, 0, stream>>>(W3,  Wt3,  H2v, P);
    k_wt_bf16<<<(P   * P   + 255) / 256, 256, 0, stream>>>(Wsp, Wtsp, P,   P);
    k_wt_bf16<<<(P   * P   + 255) / 256, 256, 0, stream>>>(Wdp, Wtdp, P,   P);

    // MLP: h1 = relu(X@W1+b1), h2 = relu(h1@W2+b2), h = h2@W3+b3 (-> out[:,0:64] + bf16 copy)
    k_gemm_bf16<<<dim3(mblocks, H1v / BN), 256, 0, stream>>>(Xbf, Wt1, b1, nullptr, 0, h1,  Nn, H1v, IN,  1);
    k_gemm_bf16<<<dim3(mblocks, H2v / BN), 256, 0, stream>>>(h1,  Wt2, b2, nullptr, 0, h2,  Nn, H2v, H1v, 1);
    k_gemm_bf16<<<dim3(mblocks, P   / BN), 256, 0, stream>>>(h2,  Wt3, b3, out, 2 * P, hbf, Nn, P,   H2v, 0);
    // node-level projections (instead of per-edge): hs = h@Wsp+bsp, hd = h@Wdp+bdp
    k_gemm_bf16<<<dim3(mblocks, P / BN), 256, 0, stream>>>(hbf, Wtsp, bsp, hs, P, nullptr, Nn, P, P, 0);
    k_gemm_bf16<<<dim3(mblocks, P / BN), 256, 0, stream>>>(hbf, Wtdp, bdp, hd, P, nullptr, Nn, P, P, 0);
    // zero agg half, then edge gather/gate/scatter
    {
        int n2 = Nn * 32;
        k_zero_agg<<<(n2 + 255) / 256, 256, 0, stream>>>(out, n2);
    }
    k_edges<<<(E + 7) / 8, 256, 0, stream>>>(hs, hd, src, dst, out, E);
}
